// MP_JEPA_77756087927082
// MI455X (gfx1250) — matrix operand order repaired and checked by reading, not compiled, not run on hardware
//
#include <hip/hip_runtime.h>

#define N_NODES 100000
#define N_EDGES 1600000
#define DD      128
#define PE_DIM  4
#define T_TGT   4096
#define P_PAIRS 131072
#define H_PRED  256
#define D_OUT   128

typedef __attribute__((ext_vector_type(16))) __bf16 v16bf;
typedef __attribute__((ext_vector_type(8)))  float  v8f;
typedef unsigned short u16;
typedef unsigned int   u32;

union V16BF { v16bf v; u16 u[16]; uint4 q[2]; };

__device__ __forceinline__ u16 f2bf(float f) {
    u32 u = __float_as_uint(f);
    u32 r = u + 0x7FFFu + ((u >> 16) & 1u);   // round-to-nearest-even
    return (u16)(r >> 16);
}
__device__ __forceinline__ float bf2f(u16 b) {
    return __uint_as_float(((u32)b) << 16);
}

// ---- A fragment: 16x32 bf16, ISA 7.12.2 layout.
// lane = 16*h + r holds row M=r, elements i<8 -> k = k0+8h+i ; i>=8 -> k = k0+16+8h+(i-8)
__device__ __forceinline__ v16bf load_a_bf(const u16* __restrict__ A, long long row,
                                           int K, int k0, int h) {
    V16BF f;
    f.q[0] = *(const uint4*)(A + row * K + k0 + 8 * h);
    f.q[1] = *(const uint4*)(A + row * K + k0 + 16 + 8 * h);
    return f.v;
}
__device__ __forceinline__ v16bf load_a_f32(const float* __restrict__ A, long long row,
                                            int K, int k0, int h) {
    const float4* p0 = (const float4*)(A + row * K + k0 + 8 * h);
    const float4* p1 = (const float4*)(A + row * K + k0 + 16 + 8 * h);
    float4 a = p0[0], b = p0[1], c = p1[0], d = p1[1];
    V16BF f;
    f.u[0]=f2bf(a.x);  f.u[1]=f2bf(a.y);  f.u[2]=f2bf(a.z);  f.u[3]=f2bf(a.w);
    f.u[4]=f2bf(b.x);  f.u[5]=f2bf(b.y);  f.u[6]=f2bf(b.z);  f.u[7]=f2bf(b.w);
    f.u[8]=f2bf(c.x);  f.u[9]=f2bf(c.y);  f.u[10]=f2bf(c.z); f.u[11]=f2bf(c.w);
    f.u[12]=f2bf(d.x); f.u[13]=f2bf(d.y); f.u[14]=f2bf(d.z); f.u[15]=f2bf(d.w);
    return f.v;
}
// ---- B fragment: pre-packed, 1KB per fragment, per-lane 32B contiguous.
__device__ __forceinline__ v16bf load_b(const u16* __restrict__ Bp, long long frag, int lane) {
    const uint4* p = (const uint4*)(Bp + frag * 512 + lane * 16);
    V16BF f; f.q[0] = p[0]; f.q[1] = p[1]; return f.v;
}

// Pre-pack B (K x Nn, row-major fp32) into WMMA-fragment order (bf16).
// B 32x16 bf16 layout: lane = 16*h + r, element e<8 -> K=k0+8h+e ; e>=8 -> K=k0+16+8h+(e-8); col = n0 + (lane&15).
__global__ void prepack_b(const float* __restrict__ B, u16* __restrict__ Bp, int K, int Nn) {
    int id = blockIdx.x * blockDim.x + threadIdx.x;
    int total = (K / 32) * (Nn / 16) * 512;
    if (id >= total) return;
    int e    = id & 15;
    int lane = (id >> 4) & 31;
    int frag = id >> 9;
    int NT = Nn / 16;
    int kt = frag / NT, nt = frag % NT;
    int h = lane >> 4, r = lane & 15;
    int k   = kt * 32 + (e < 8 ? 8 * h + e : 16 + 8 * h + (e - 8));
    int col = nt * 16 + r;
    Bp[(long long)frag * 512 + lane * 16 + e] = f2bf(B[(long long)k * Nn + col]);
}

// ------------------- fused WMMA GEMM -------------------
// C[M x Nn] = A[M x K] @ B ; A rows optionally gathered through gidx.
// epi 0: store bf16.  epi 1: +=bias[col] + pep[row][0..7]·pwr[q][col], ReLU, store bf16.
// epi 2: (+bias[col]) * mask[row], store fp32.
__global__ __launch_bounds__(128) void wmma_gemm_kernel(
    const float* __restrict__ Af, const u16* __restrict__ Ab, const int* __restrict__ gidx,
    const u16* __restrict__ Bp, int M, int K, int Nn,
    u16* __restrict__ Cb, float* __restrict__ Cf, int epi,
    const float* __restrict__ ebias, const float* __restrict__ pep,
    const float* __restrict__ pwr, const float* __restrict__ mask)
{
    const int lane = threadIdx.x & 31;
    const int wave = threadIdx.x >> 5;
    const int h = lane >> 4, r = lane & 15;
    const int m0 = (blockIdx.x * 4 + wave) * 16;
    if (m0 >= M) return;                     // wave-uniform: EXEC stays full for WMMA
    const int ntb = blockIdx.y * 8;
    const int NT = Nn >> 4;
    const int KT = K >> 5;

    int ar = m0 + r;
    if (ar > M - 1) ar = M - 1;              // row clamp is safe: A row only feeds same C row
    if (gidx) ar = gidx[ar];
    const long long arow = ar;

    v8f acc[8] = {};
    for (int kt = 0; kt < KT; ++kt) {
        const int k0 = kt << 5;
        v16bf a = Af ? load_a_f32(Af, arow, K, k0, h)
                     : load_a_bf (Ab, arow, K, k0, h);
        // Preload all 8 B fragments into distinct registers so the compiler can
        // clause the 16 b128 loads and stagger s_wait_loadcnt across the WMMAs.
        v16bf b[8];
#pragma unroll
        for (int nt = 0; nt < 8; ++nt)
            b[nt] = load_b(Bp, (long long)kt * NT + (ntb + nt), lane);
#pragma unroll
        for (int nt = 0; nt < 8; ++nt)
            acc[nt] = __builtin_amdgcn_wmma_f32_16x16x32_bf16(
                false, a, false, b[nt], (short)0, acc[nt], false, false);
    }

#pragma unroll
    for (int nt = 0; nt < 8; ++nt) {
        const int col = (ntb + nt) * 16 + r;
        float bias = 0.0f;
        if (epi != 0) bias = ebias[col];
        float wr[8];
        if (epi == 1) {
#pragma unroll
            for (int q = 0; q < 8; ++q) wr[q] = pwr[q * Nn + col];
        }
#pragma unroll
        for (int j = 0; j < 8; ++j) {
            const int row = m0 + j + 8 * h;
            if (row < M) {
                float v = acc[nt][j];
                if (epi == 1) {
                    const float* pp = pep + (long long)row * 8;
                    v += bias;
#pragma unroll
                    for (int q = 0; q < 8; ++q) v += pp[q] * wr[q];
                    v = fmaxf(v, 0.0f);
                    Cb[(long long)row * Nn + col] = f2bf(v);
                } else if (epi == 2) {
                    v = (v + bias) * mask[row];
                    Cf[(long long)row * Nn + col] = v;
                } else {
                    Cb[(long long)row * Nn + col] = f2bf(v);
                }
            }
        }
    }
}

// ------------------- graph kernels -------------------
__global__ void deg_kernel(const int* __restrict__ dst, float* __restrict__ deg, int E) {
    int e = blockIdx.x * blockDim.x + threadIdx.x;
    if (e >= E) return;
    __hip_atomic_fetch_add(deg + dst[e], 1.0f, __ATOMIC_RELAXED, __HIP_MEMORY_SCOPE_AGENT);
}

// one wave per edge; each lane scatters 4 features (bf16 in, fp32 atomic accumulate)
__global__ void agg_kernel(const u16* __restrict__ Xb, const int* __restrict__ src,
                           const int* __restrict__ dst, float* __restrict__ acc, int E) {
    int tid = blockIdx.x * blockDim.x + threadIdx.x;
    int e = tid >> 5;
    if (e >= E) return;
    int lane = tid & 31;
    int d = lane * 4;
    int s = src[e], dd = dst[e];
    union { uint2 q; u16 u[4]; } v;
    v.q = *(const uint2*)(Xb + (long long)s * DD + d);
    float* out = acc + (long long)dd * DD + d;
#pragma unroll
    for (int i = 0; i < 4; ++i)
        __hip_atomic_fetch_add(out + i, bf2f(v.u[i]), __ATOMIC_RELAXED, __HIP_MEMORY_SCOPE_AGENT);
}

__global__ void norm_kernel(const float* __restrict__ acc, const float* __restrict__ deg,
                            u16* __restrict__ out, int relu, int n_nodes) {
    int i = blockIdx.x * blockDim.x + threadIdx.x;
    if (i >= n_nodes * DD) return;
    int n = i >> 7;
    float dg = deg[n]; if (dg < 1.0f) dg = 1.0f;
    float v = acc[i] / dg;
    if (relu) v = fmaxf(v, 0.0f);
    out[i] = f2bf(v);
}

__global__ void tgt_kernel(const float* __restrict__ acc, const float* __restrict__ deg,
                           const int* __restrict__ tgt, float* __restrict__ out, int T) {
    int i = blockIdx.x * blockDim.x + threadIdx.x;
    if (i >= T * DD) return;
    int t = i >> 7, d = i & 127;
    int n = tgt[t];
    float dg = deg[n]; if (dg < 1.0f) dg = 1.0f;
    out[i] = acc[(long long)n * DD + d] / dg;
}

// pe = pos_enc @ Wpe.T  (N x 4)
__global__ void pe_kernel(const float* __restrict__ pos, const float* __restrict__ Wpe,
                          float* __restrict__ pe, int n_nodes) {
    int i = blockIdx.x * blockDim.x + threadIdx.x;
    if (i >= n_nodes * PE_DIM) return;
    int n = i >> 2, o = i & 3;
    float s = 0.0f;
#pragma unroll
    for (int j = 0; j < 4; ++j) s += pos[n * 4 + j] * Wpe[o * 4 + j];
    pe[i] = s;
}

// pep[p][0..3] = pe[pair_s[p]], pep[p][4..7] = pe[target_nodes[pair_t[p]]]
__global__ void pep_kernel(const float* __restrict__ pe, const int* __restrict__ ps,
                           const int* __restrict__ pt, const int* __restrict__ tgt,
                           float* __restrict__ pep, int P) {
    int i = blockIdx.x * blockDim.x + threadIdx.x;
    if (i >= P * 8) return;
    int p = i >> 3, j = i & 7;
    int n = (j < 4) ? ps[p] : tgt[pt[p]];
    pep[i] = pe[n * 4 + (j & 3)];
}

// eb1 = bp1 + z @ Wp1[128:192, :]   (z is broadcast over all pairs -> fold into bias)
__global__ void bias1_kernel(const float* __restrict__ bp1, const float* __restrict__ z,
                             const float* __restrict__ Wp1, float* __restrict__ eb) {
    int c = threadIdx.x;
    if (c >= H_PRED) return;
    float s = bp1[c];
    for (int j = 0; j < 64; ++j) s += z[j] * Wp1[(128 + j) * H_PRED + c];
    eb[c] = s;
}

extern "C" void kernel_launch(void* const* d_in, const int* in_sizes, int n_in,
                              void* d_out, int out_size, void* d_ws, size_t ws_size,
                              hipStream_t stream)
{
    const float* x    = (const float*)d_in[0];
    const float* mx   = (const float*)d_in[1];
    const float* pos  = (const float*)d_in[2];
    const int*   esrc = (const int*)d_in[3];
    const int*   edst = (const int*)d_in[4];
    const int*   tgt  = (const int*)d_in[5];
    const int*   pt   = (const int*)d_in[6];
    const int*   ps   = (const int*)d_in[7];
    const float* pmask= (const float*)d_in[8];
    const float* W1t  = (const float*)d_in[9];
    const float* W2t  = (const float*)d_in[10];
    const float* W1c  = (const float*)d_in[11];
    const float* W2c  = (const float*)d_in[12];
    const float* Wpe  = (const float*)d_in[13];
    const float* z    = (const float*)d_in[14];
    const float* Wp1  = (const float*)d_in[15];
    const float* bp1  = (const float*)d_in[16];
    const float* Wp2  = (const float*)d_in[17];
    const float* bp2  = (const float*)d_in[18];
    (void)in_sizes; (void)n_in; (void)out_size; (void)ws_size;

    char* w = (char*)d_ws;
    size_t off = 0;
    auto alloc = [&](size_t b) -> char* {
        char* p = w + off;
        off = (off + b + 255) & ~(size_t)255;
        return p;
    };
    float* deg   = (float*)alloc((size_t)N_NODES * 4);
    float* acc   = (float*)alloc((size_t)N_NODES * DD * 4);
    u16*   xw    = (u16*)  alloc((size_t)N_NODES * DD * 2);
    u16*   hb    = (u16*)  alloc((size_t)N_NODES * DD * 2);
    u16*   ctx   = (u16*)  alloc((size_t)N_NODES * DD * 2);
    float* pe    = (float*)alloc((size_t)N_NODES * PE_DIM * 4);
    float* pep   = (float*)alloc((size_t)P_PAIRS * 8 * 4);
    float* eb1   = (float*)alloc(H_PRED * 4);
    u16* bp_w1t  = (u16*)alloc(32768);
    u16* bp_w2t  = (u16*)alloc(32768);
    u16* bp_w1c  = (u16*)alloc(32768);
    u16* bp_w2c  = (u16*)alloc(32768);
    u16* bp_wp1  = (u16*)alloc(65536);
    u16* bp_wp2  = (u16*)alloc(65536);
    u16* hidden  = (u16*)alloc((size_t)P_PAIRS * H_PRED * 2);

    float* pred_out = (float*)d_out;                              // [P_PAIRS x 128]
    float* temb_out = (float*)d_out + (size_t)P_PAIRS * D_OUT;    // [T x 128]

    const dim3 blk(128, 1, 1);
    auto gx = [](int M) { return (unsigned)((M + 63) / 64); };

    // degree
    hipMemsetAsync(deg, 0, (size_t)N_NODES * 4, stream);
    deg_kernel<<<(N_EDGES + 255) / 256, 256, 0, stream>>>(edst, deg, N_EDGES);

    // weight pre-pack into WMMA fragment order
    prepack_b<<<(4 * 8 * 512 + 255) / 256, 256, 0, stream>>>(W1t, bp_w1t, 128, 128);
    prepack_b<<<(4 * 8 * 512 + 255) / 256, 256, 0, stream>>>(W2t, bp_w2t, 128, 128);
    prepack_b<<<(4 * 8 * 512 + 255) / 256, 256, 0, stream>>>(W1c, bp_w1c, 128, 128);
    prepack_b<<<(4 * 8 * 512 + 255) / 256, 256, 0, stream>>>(W2c, bp_w2c, 128, 128);
    prepack_b<<<(4 * 16 * 512 + 255) / 256, 256, 0, stream>>>(Wp1, bp_wp1, 128, 256); // rows 0..127
    prepack_b<<<(8 * 8 * 512 + 255) / 256, 256, 0, stream>>>(Wp2, bp_wp2, 256, 128);

    // positional encodings + folded bias
    pe_kernel<<<(N_NODES * 4 + 255) / 256, 256, 0, stream>>>(pos, Wpe, pe, N_NODES);
    pep_kernel<<<(P_PAIRS * 8 + 255) / 256, 256, 0, stream>>>(pe, ps, pt, tgt, pep, P_PAIRS);
    bias1_kernel<<<1, 256, 0, stream>>>(bp1, z, Wp1, eb1);

    const unsigned agg_grid = (unsigned)(((long long)N_EDGES * 32 + 255) / 256);

    // -------- target GCN (x, W1t/W2t) --------
    wmma_gemm_kernel<<<dim3(gx(N_NODES), 1), blk, 0, stream>>>(
        x, nullptr, nullptr, bp_w1t, N_NODES, 128, 128, xw, nullptr, 0,
        nullptr, nullptr, nullptr, nullptr);
    hipMemsetAsync(acc, 0, (size_t)N_NODES * DD * 4, stream);
    agg_kernel<<<agg_grid, 256, 0, stream>>>(xw, esrc, edst, acc, N_EDGES);
    norm_kernel<<<(N_NODES * DD + 255) / 256, 256, 0, stream>>>(acc, deg, hb, 1, N_NODES);
    wmma_gemm_kernel<<<dim3(gx(N_NODES), 1), blk, 0, stream>>>(
        nullptr, hb, nullptr, bp_w2t, N_NODES, 128, 128, xw, nullptr, 0,
        nullptr, nullptr, nullptr, nullptr);
    hipMemsetAsync(acc, 0, (size_t)N_NODES * DD * 4, stream);
    agg_kernel<<<agg_grid, 256, 0, stream>>>(xw, esrc, edst, acc, N_EDGES);
    tgt_kernel<<<(T_TGT * DD + 255) / 256, 256, 0, stream>>>(acc, deg, tgt, temb_out, T_TGT);

    // -------- context GCN (masked_x, W1c/W2c) --------
    wmma_gemm_kernel<<<dim3(gx(N_NODES), 1), blk, 0, stream>>>(
        mx, nullptr, nullptr, bp_w1c, N_NODES, 128, 128, xw, nullptr, 0,
        nullptr, nullptr, nullptr, nullptr);
    hipMemsetAsync(acc, 0, (size_t)N_NODES * DD * 4, stream);
    agg_kernel<<<agg_grid, 256, 0, stream>>>(xw, esrc, edst, acc, N_EDGES);
    norm_kernel<<<(N_NODES * DD + 255) / 256, 256, 0, stream>>>(acc, deg, hb, 1, N_NODES);
    wmma_gemm_kernel<<<dim3(gx(N_NODES), 1), blk, 0, stream>>>(
        nullptr, hb, nullptr, bp_w2c, N_NODES, 128, 128, xw, nullptr, 0,
        nullptr, nullptr, nullptr, nullptr);
    hipMemsetAsync(acc, 0, (size_t)N_NODES * DD * 4, stream);
    agg_kernel<<<agg_grid, 256, 0, stream>>>(xw, esrc, edst, acc, N_EDGES);
    norm_kernel<<<(N_NODES * DD + 255) / 256, 256, 0, stream>>>(acc, deg, ctx, 0, N_NODES);

    // -------- predictor --------
    // hidden = ReLU(ctx[pair_s] @ Wp1[0:128] + pep·Wp1[192:200] + (bp1 + z@Wp1[128:192]))
    wmma_gemm_kernel<<<dim3(gx(P_PAIRS), 2), blk, 0, stream>>>(
        nullptr, ctx, ps, bp_wp1, P_PAIRS, 128, 256, hidden, nullptr, 1,
        eb1, pep, Wp1 + 192 * H_PRED, nullptr);
    // pred = (hidden @ Wp2 + bp2) * pair_mask
    wmma_gemm_kernel<<<dim3(gx(P_PAIRS), 1), blk, 0, stream>>>(
        nullptr, hidden, nullptr, bp_wp2, P_PAIRS, 256, 128, nullptr, pred_out, 2,
        bp2, nullptr, nullptr, pmask);
}
